// LibraModel_81226421502326
// MI455X (gfx1250) — matrix operand order repaired and checked by reading, not compiled, not run on hardware
//
#include <hip/hip_runtime.h>
#include <hip/hip_bf16.h>
#include <math.h>

typedef __bf16 bf16;
typedef __attribute__((ext_vector_type(16))) __bf16 v16bf;
typedef __attribute__((ext_vector_type(8)))  float  v8f;
typedef __attribute__((ext_vector_type(4)))  unsigned int v4u;
typedef __attribute__((ext_vector_type(8)))  unsigned int v8u;

#define Bsz 16
#define Sln 1024
#define INd 32
#define Hd  256
#define G4  1024   // 4*H
#define NHh 8
#define HDd 32

// ---------- WMMA fragment loaders (CDNA5 16x16x32 bf16 layouts) ----------
// A: 16x32 (MxK). lane L: m=L&15, half=L>>4.
//    elements 0..7  = bf16 @ k = kc*32 + 8*half + e        (16B)
//    elements 8..15 = bf16 @ k = kc*32 + 16 + 8*half + e-8 (16B, +32B from first)
__device__ __forceinline__ v16bf load_frag_a(const bf16* A, int ldk, int m, int kc, int half) {
  const char* p = (const char*)(A + (size_t)m * ldk + kc * 32 + 8 * half);
  union { uint4 q[2]; v16bf v; } u;
  u.q[0] = *(const uint4*)(p);
  u.q[1] = *(const uint4*)(p + 32);
  return u.v;
}
// B: 32x16 (KxN) = W^T tile, W stored (N,K) row-major. lane L: n=L&15, half=L>>4.
//    elements 0..7 = bf16 @ k = kc*32 + 8*half + e; 8..15 = bf16 @ k = kc*32 + 16 + 8*half + e-8
__device__ __forceinline__ v16bf load_frag_b(const bf16* W, int ldk, int n, int kc, int half) {
  const char* p = (const char*)(W + (size_t)n * ldk + kc * 32 + 8 * half);
  union { uint4 q[2]; v16bf v; } u;
  u.q[0] = *(const uint4*)(p);
  u.q[1] = *(const uint4*)(p + 32);
  return u.v;
}
__device__ __forceinline__ v8f wmma_bf16(v16bf a, v16bf b, v8f c) {
  return __builtin_amdgcn_wmma_f32_16x16x32_bf16(false, a, false, b, (short)0, c, false, false);
}

__device__ __forceinline__ float fsigmoid(float x) { return 1.0f / (1.0f + __expf(-x)); }
__device__ __forceinline__ float ftanh(float x) {
  float e = __expf(2.0f * x);
  return (e - 1.0f) / (e + 1.0f);
}

// ---------- Tensor Data Mover: 2D strided tile -> LDS (ISA ch.8 D# layout) ----------
// group0: [1:0]=count=1 | [63:32]=lds_addr | [120:64]=global byte addr | [127:126]=type 2
// group1: [17:16]=data_size(2=4B) | [79:48]=tensor_dim0 | [111:80]=tensor_dim1 |
//         [127:112]=tile_dim0 | [143:128]=tile_dim1 | [207:160]=tensor_dim0_stride
__device__ __forceinline__ void tdm_load_tile_2d(unsigned int lds_byte_off, const void* gptr,
                                                 unsigned int tensor_d0, unsigned int tensor_d1,
                                                 unsigned int tile_d0, unsigned int tile_d1,
                                                 unsigned long long row_stride_elems) {
  unsigned long long ga = (unsigned long long)(__UINTPTR_TYPE__)gptr;
  v4u g0;
  g0[0] = 1u;                                                        // count=1, user descriptor
  g0[1] = lds_byte_off;
  g0[2] = (unsigned int)(ga & 0xFFFFFFFFull);
  g0[3] = (unsigned int)((ga >> 32) & 0x01FFFFFFull) | (2u << 30);   // type=2 ("image")
  v8u g1;
  g1[0] = 0x00020000u;                                               // data_size = 4 bytes
  g1[1] = (tensor_d0 & 0xFFFFu) << 16;
  g1[2] = (tensor_d0 >> 16) | ((tensor_d1 & 0xFFFFu) << 16);
  g1[3] = (tensor_d1 >> 16) | (tile_d0 << 16);
  g1[4] = tile_d1 & 0xFFFFu;                                         // tile_dim2 = 0 (2D)
  g1[5] = (unsigned int)(row_stride_elems & 0xFFFFFFFFull);
  g1[6] = (unsigned int)((row_stride_elems >> 32) & 0xFFFFull);      // dim1_stride unused (2D)
  g1[7] = 0u;
  asm volatile("tensor_load_to_lds %0, %1" :: "s"(g0), "s"(g1) : "memory");
}

// ---------- f32 -> bf16 conversion ----------
__global__ void cvt_bf16_kernel(const float* __restrict__ src, bf16* __restrict__ dst, int n) {
  int i = blockIdx.x * blockDim.x + threadIdx.x;
  int stride = gridDim.x * blockDim.x;
  for (; i < n; i += stride) dst[i] = (bf16)src[i];
}

// ---------- Generic bf16 WMMA GEMM: Y(MxN,f32) = A(MxK) @ W(N,K)^T + bias1 + bias2 ----------
__global__ void __launch_bounds__(256)
gemm_bf16_kernel(const bf16* __restrict__ A, const bf16* __restrict__ W,
                 const float* __restrict__ bias1, const float* __restrict__ bias2,
                 float* __restrict__ Y, int M, int N, int K) {
  int lane = threadIdx.x & 31, wave = threadIdx.x >> 5;
  int half = lane >> 4, l15 = lane & 15;
  int nT = N >> 4;
  int tile = blockIdx.x * 8 + wave;
  int mt = tile / nT, nt = tile - mt * nT;
  int arow = mt * 16 + l15;
  int n = nt * 16 + l15;
  v8f acc = {};
  int KC = K >> 5;
  for (int kc = 0; kc < KC; ++kc) {
    v16bf a = load_frag_a(A, K, arow, kc, half);
    v16bf b = load_frag_b(W, K, n, kc, half);
    acc = wmma_bf16(a, b, acc);
  }
  float bb = (bias1 ? bias1[n] : 0.0f) + (bias2 ? bias2[n] : 0.0f);
#pragma unroll
  for (int r = 0; r < 8; ++r) {
    int mm = mt * 16 + r + 8 * half;
    Y[(size_t)mm * N + n] = acc[r] + bb;
  }
}

// ---------- LSTM recurrence: single workgroup, 32 waves, 1024 sequential steps ----------
// gates(16x1024) = xproj_slab(t) + h(16x256) @ Whh(1024x256)^T
// xproj slabs (64KB/step) are staged one step ahead by the TDM into a double buffer.
// LDS: xp[2][16][1024]f32 (128K) | gates 16x1024 f32 (64K) | c 16x256 f32 (16K) | h bf16 (8K)
__global__ void __launch_bounds__(1024, 1)
lstm_recur_kernel(const float* __restrict__ xproj, const bf16* __restrict__ Whh,
                  bf16* __restrict__ h_out) {
  extern __shared__ char smem[];
  float* xp_lds = (float*)smem;                       // 2 x 65536 B
  float* g_lds  = (float*)(smem + 131072);            // 65536 B
  float* c_lds  = (float*)(smem + 131072 + 65536);    // 16384 B
  bf16*  h_lds  = (bf16*) (smem + 131072 + 65536 + 16384); // 8192 B

  int tid  = threadIdx.x;
  int lane = tid & 31, wave = tid >> 5;
  int half = lane >> 4, l15 = lane & 15;
  int n0a = (wave * 2) * 16;          // this wave's two N-tiles
  int n0b = n0a + 16;
  unsigned int xp_off = (unsigned int)(__UINTPTR_TYPE__)xp_lds;  // LDS byte offset (addr[31:0])

  for (int i = tid; i < Bsz * Hd; i += 1024) { c_lds[i] = 0.0f; h_lds[i] = (bf16)0.0f; }
  // warm W_hh rows for this wave into the cache hierarchy (global_prefetch_b8)
  __builtin_prefetch(Whh + (size_t)(n0a + lane) * Hd, 0, 3);
  if (wave == 0) {
    // stage slab t=0 into buffer 0
    tdm_load_tile_2d(xp_off, xproj, G4, Bsz, G4, Bsz, (unsigned long long)Sln * G4);
  }
  __syncthreads();

  for (int t = 0; t < Sln; ++t) {
    if (wave == 0) __builtin_amdgcn_s_wait_tensorcnt(0);   // slab t landed in LDS
    __syncthreads();                                       // visible to all; old buffer drained
    if (wave == 0 && t + 1 < Sln) {
      tdm_load_tile_2d(xp_off + ((unsigned int)((t + 1) & 1)) * 65536u,
                       xproj + (size_t)(t + 1) * G4,
                       G4, Bsz, G4, Bsz, (unsigned long long)Sln * G4);
    }
    const float* xpb = xp_lds + (size_t)(t & 1) * (Bsz * G4);

    // A fragments (shared h) from LDS, kept in registers for both N-tiles
    v16bf afr[8];
#pragma unroll
    for (int kc = 0; kc < 8; ++kc) afr[kc] = load_frag_a(h_lds, Hd, l15, kc, half);

    v8f acc0 = {}, acc1 = {};
#pragma unroll
    for (int kc = 0; kc < 8; ++kc)
      acc0 = wmma_bf16(afr[kc], load_frag_b(Whh, Hd, n0a + l15, kc, half), acc0);
#pragma unroll
    for (int kc = 0; kc < 8; ++kc)
      acc1 = wmma_bf16(afr[kc], load_frag_b(Whh, Hd, n0b + l15, kc, half), acc1);

    // add TDM-staged xproj, activate (quarter 2 = g -> tanh; i/f/o -> sigmoid)
#pragma unroll
    for (int which = 0; which < 2; ++which) {
      v8f acc = which ? acc1 : acc0;
      int n = (which ? n0b : n0a) + l15;
      int quarter = n >> 8;
#pragma unroll
      for (int r = 0; r < 8; ++r) {
        int m = r + 8 * half;  // batch row
        float v = acc[r] + xpb[m * G4 + n];
        v = (quarter == 2) ? ftanh(v) : fsigmoid(v);
        g_lds[m * G4 + n] = v;
      }
    }
    __syncthreads();

    // c/h elementwise update: 4096 (b,j) pairs over 1024 threads
#pragma unroll
    for (int i = 0; i < 4; ++i) {
      int p = tid + i * 1024;
      int b = p >> 8, j = p & 255;
      float iv = g_lds[b * G4 + j];
      float fv = g_lds[b * G4 + 256 + j];
      float gv = g_lds[b * G4 + 512 + j];
      float ov = g_lds[b * G4 + 768 + j];
      float c  = fv * c_lds[b * Hd + j] + iv * gv;
      c_lds[b * Hd + j] = c;
      float h = ov * ftanh(c);
      h_lds[b * Hd + j] = (bf16)h;
      h_out[((size_t)b * Sln + t) * Hd + j] = (bf16)h;
    }
    // top-of-loop barrier of next iteration orders h_lds writes vs A-frag reads
  }
}

// ---------- Q projection for the last time step only ----------
__global__ void qproj_last_kernel(const bf16* __restrict__ h1, const float* __restrict__ Wq,
                                  const float* __restrict__ bq, float* __restrict__ qout) {
  int b = blockIdx.x, n = threadIdx.x;
  const bf16* hp = h1 + ((size_t)b * Sln + (Sln - 1)) * Hd;
  float s = bq[n];
  for (int j = 0; j < Hd; ++j) s += (float)hp[j] * Wq[n * Hd + j];
  qout[b * Hd + n] = s;
}

// ---------- single-query decay attention (per (batch, head) block) ----------
__global__ void __launch_bounds__(256)
attn_last_kernel(const float* __restrict__ q, const float* __restrict__ kall,
                 const float* __restrict__ vall, float* __restrict__ attn) {
  __shared__ float sc[Sln];
  __shared__ float red[256];
  __shared__ float qs[HDd];
  __shared__ float part[8][HDd];
  int b = blockIdx.x >> 3, hd = blockIdx.x & 7;
  int tid = threadIdx.x;
  if (tid < HDd) qs[tid] = q[b * Hd + hd * HDd + tid];
  __syncthreads();

  const float lnDecay = -0.05129329438755058f;   // ln(0.95)
  const float invSqrt = 0.17677669529663687f;    // 1/sqrt(32)
  float lmax = -1e30f;
#pragma unroll
  for (int i = 0; i < 4; ++i) {
    int t = i * 256 + tid;
    const float* kp = kall + ((size_t)b * Sln + t) * Hd + hd * HDd;
    float s = 0.0f;
    for (int d = 0; d < HDd; ++d) s += qs[d] * kp[d];
    s = s * invSqrt * __expf(lnDecay * (float)(Sln - 1 - t));
    sc[t] = s;
    lmax = fmaxf(lmax, s);
  }
  red[tid] = lmax; __syncthreads();
  for (int w = 128; w > 0; w >>= 1) { if (tid < w) red[tid] = fmaxf(red[tid], red[tid + w]); __syncthreads(); }
  float mx = red[0];
  __syncthreads();
  float lsum = 0.0f;
#pragma unroll
  for (int i = 0; i < 4; ++i) { int t = i * 256 + tid; float p = __expf(sc[t] - mx); sc[t] = p; lsum += p; }
  red[tid] = lsum; __syncthreads();
  for (int w = 128; w > 0; w >>= 1) { if (tid < w) red[tid] += red[tid + w]; __syncthreads(); }
  float inv = 1.0f / red[0];

  int d = tid & 31, chunk = tid >> 5;
  float acc = 0.0f;
  for (int t = chunk * 128; t < chunk * 128 + 128; ++t)
    acc += sc[t] * vall[((size_t)b * Sln + t) * Hd + hd * HDd + d];
  part[chunk][d] = acc;
  __syncthreads();
  if (tid < HDd) {
    float o = 0.0f;
#pragma unroll
    for (int c2 = 0; c2 < 8; ++c2) o += part[c2][tid];
    attn[b * Hd + hd * HDd + tid] = o * inv;
  }
}

// ---------- tail: context = attn @ Wo^T + bo ; mean / log_var heads ----------
__global__ void __launch_bounds__(256)
tail_kernel(const float* __restrict__ attn, const float* __restrict__ Wo, const float* __restrict__ bo,
            const float* __restrict__ Wm, const float* __restrict__ bm,
            const float* __restrict__ Wv, const float* __restrict__ bvv,
            float* __restrict__ out) {
  __shared__ float ctx[Bsz * Hd];
  int tid = threadIdx.x;
  for (int i = 0; i < 16; ++i) {
    int p = tid + i * 256; int b = p >> 8, n = p & 255;
    float s = bo[n];
    const float* ap = attn + b * Hd;
    for (int j = 0; j < Hd; ++j) s += ap[j] * Wo[n * Hd + j];
    ctx[p] = s;
  }
  __syncthreads();
  if (tid < 160) {
    int which = tid / 80, p = tid % 80;
    int b = p / 5, n = p % 5;
    const float* W = which ? Wv : Wm;
    const float* bias = which ? bvv : bm;
    float s = bias[n];
    for (int j = 0; j < Hd; ++j) s += ctx[b * Hd + j] * W[n * Hd + j];
    out[which * 80 + p] = s;   // [0:80)=mean, [80:160)=log_var
  }
}

extern "C" void kernel_launch(void* const* d_in, const int* in_sizes, int n_in,
                              void* d_out, int out_size, void* d_ws, size_t ws_size,
                              hipStream_t stream) {
  (void)in_sizes; (void)n_in; (void)out_size; (void)ws_size;
  const float* x     = (const float*)d_in[0];
  const float* Wih0  = (const float*)d_in[1];
  const float* Whh0  = (const float*)d_in[2];
  const float* bih0  = (const float*)d_in[3];
  const float* bhh0  = (const float*)d_in[4];
  const float* Wih1  = (const float*)d_in[5];
  const float* Whh1  = (const float*)d_in[6];
  const float* bih1  = (const float*)d_in[7];
  const float* bhh1  = (const float*)d_in[8];
  const float* Wq    = (const float*)d_in[9];
  const float* bq    = (const float*)d_in[10];
  const float* Wk    = (const float*)d_in[11];
  const float* bk    = (const float*)d_in[12];
  const float* Wv    = (const float*)d_in[13];
  const float* bv    = (const float*)d_in[14];
  const float* Wo    = (const float*)d_in[15];
  const float* bo    = (const float*)d_in[16];
  const float* Wm    = (const float*)d_in[17];
  const float* bm    = (const float*)d_in[18];
  const float* Wvar  = (const float*)d_in[19];
  const float* bvar  = (const float*)d_in[20];
  float* out = (float*)d_out;

  // workspace carve-up (256B aligned)
  char* ws = (char*)d_ws;
  size_t o = 0;
  auto carve = [&](size_t bytes) { size_t r = o; o = (o + bytes + 255) & ~(size_t)255; return r; };
  bf16*  x_bf    = (bf16*) (ws + carve((size_t)Bsz * Sln * INd * 2));
  bf16*  Wih0_bf = (bf16*) (ws + carve((size_t)G4 * INd * 2));
  bf16*  Whh0_bf = (bf16*) (ws + carve((size_t)G4 * Hd * 2));
  bf16*  Wih1_bf = (bf16*) (ws + carve((size_t)G4 * Hd * 2));
  bf16*  Whh1_bf = (bf16*) (ws + carve((size_t)G4 * Hd * 2));
  bf16*  Wk_bf   = (bf16*) (ws + carve((size_t)Hd * Hd * 2));
  bf16*  Wv_bf   = (bf16*) (ws + carve((size_t)Hd * Hd * 2));
  float* xproj   = (float*)(ws + carve((size_t)Bsz * Sln * G4 * 4));  // reused by both layers
  bf16*  h0_bf   = (bf16*) (ws + carve((size_t)Bsz * Sln * Hd * 2));
  bf16*  h1_bf   = (bf16*) (ws + carve((size_t)Bsz * Sln * Hd * 2));
  float* k_all   = (float*)(ws + carve((size_t)Bsz * Sln * Hd * 4));
  float* v_all   = (float*)(ws + carve((size_t)Bsz * Sln * Hd * 4));
  float* q_ws    = (float*)(ws + carve((size_t)Bsz * Hd * 4));
  float* a_ws    = (float*)(ws + carve((size_t)Bsz * Hd * 4));

  auto cvt = [&](const float* s, bf16* d, int n) {
    int blk = (n + 255) / 256; if (blk > 2048) blk = 2048;
    cvt_bf16_kernel<<<blk, 256, 0, stream>>>(s, d, n);
  };
  cvt(x,    x_bf,    Bsz * Sln * INd);
  cvt(Wih0, Wih0_bf, G4 * INd);
  cvt(Whh0, Whh0_bf, G4 * Hd);
  cvt(Wih1, Wih1_bf, G4 * Hd);
  cvt(Whh1, Whh1_bf, G4 * Hd);
  cvt(Wk,   Wk_bf,   Hd * Hd);
  cvt(Wv,   Wv_bf,   Hd * Hd);

  const int M = Bsz * Sln;                                     // 16384
  const size_t recur_lds = 131072 + 65536 + 16384 + 8192;      // xp dbl-buf + gates + c + h

  // layer 0: x_proj then recurrence
  gemm_bf16_kernel<<<(M / 16) * (G4 / 16) / 8, 256, 0, stream>>>(
      x_bf, Wih0_bf, bih0, bhh0, xproj, M, G4, INd);
  lstm_recur_kernel<<<1, 1024, recur_lds, stream>>>(xproj, Whh0_bf, h0_bf);

  // layer 1: x_proj then recurrence
  gemm_bf16_kernel<<<(M / 16) * (G4 / 16) / 8, 256, 0, stream>>>(
      h0_bf, Wih1_bf, bih1, bhh1, xproj, M, G4, Hd);
  lstm_recur_kernel<<<1, 1024, recur_lds, stream>>>(xproj, Whh1_bf, h1_bf);

  // K / V projections (full sequence), Q only for the last step
  gemm_bf16_kernel<<<(M / 16) * (Hd / 16) / 8, 256, 0, stream>>>(
      h1_bf, Wk_bf, bk, nullptr, k_all, M, Hd, Hd);
  gemm_bf16_kernel<<<(M / 16) * (Hd / 16) / 8, 256, 0, stream>>>(
      h1_bf, Wv_bf, bv, nullptr, v_all, M, Hd, Hd);
  qproj_last_kernel<<<Bsz, Hd, 0, stream>>>(h1_bf, Wq, bq, q_ws);

  // decay attention at the last query position, then heads
  attn_last_kernel<<<Bsz * NHh, 256, 0, stream>>>(q_ws, k_all, v_all, a_ws);
  tail_kernel<<<1, 256, 0, stream>>>(a_ws, Wo, bo, Wm, bm, Wvar, bvar, out);
}
